// FastStackedSAE_82411832476228
// MI455X (gfx1250) — matrix operand order repaired and checked by reading, not compiled, not run on hardware
//
#include <hip/hip_runtime.h>
#include <hip/hip_bf16.h>

#define B_   256
#define T_   8
#define DIN  768
#define DSAE 8192
#define K_   32

typedef __attribute__((ext_vector_type(16))) __bf16 v16bf;
typedef __attribute__((ext_vector_type(8)))  __bf16 v8bf;
typedef __attribute__((ext_vector_type(8)))  float  v8f;

__device__ __forceinline__ v16bf concat8(v8bf lo, v8bf hi) {
    return __builtin_shufflevector(lo, hi, 0, 1, 2, 3, 4, 5, 6, 7,
                                           8, 9, 10, 11, 12, 13, 14, 15);
}

// ---------------------------------------------------------------------------
// Kernel 1: encoder GEMM  pre[b,t,h] = (x[b,t,:] - b_dec[t,:]) . W_enc[t,h,:] + b_enc[t,h]
// bf16 WMMA (16x16x32), f32 accumulate. Block = 256 threads = 8 waves,
// computes a 128(M=batch) x 64(N=latent) tile for one t.
// ---------------------------------------------------------------------------
__global__ __launch_bounds__(256)
void sae_encode_wmma(const float* __restrict__ x,
                     const float* __restrict__ W_enc,
                     const float* __restrict__ b_enc,
                     const float* __restrict__ b_dec,
                     float* __restrict__ pre) {
    // A tile: [m][k] bf16, padded to 40 elems/row (80B, conflict-free b128 reads)
    __shared__ __bf16 As[128][40];
    // B tile transposed: [k][n] bf16, padded to 72 elems/row (144B)
    __shared__ __bf16 Bs[32][72];

    const int tid   = threadIdx.x;
    const int t     = blockIdx.z;
    const int nBase = blockIdx.x * 64;
    const int mBase = blockIdx.y * 128;
    const int w     = tid >> 5;
    const int lane  = tid & 31;
    const int nSub  = (w & 3) * 16;   // 4 N-subtiles across waves
    const int mWave = (w >> 2) * 64;  // 2 M-groups across waves, 4 subtiles each

    v8f acc[4] = {};

    for (int kb = 0; kb < DIN; kb += 32) {
        // ---- cooperative load A chunk: 128 rows x 32 k (x - b_dec) -> bf16
#pragma unroll
        for (int r = 0; r < 4; ++r) {
            int lin = tid + r * 256;          // 0..1023
            int m   = lin >> 3;               // 0..127
            int f4  = lin & 7;                // 0..7 (float4 index)
            const float4 xv = *(const float4*)(x + ((size_t)(mBase + m) * T_ + t) * DIN + kb + f4 * 4);
            const float4 bd = *(const float4*)(b_dec + (size_t)t * DIN + kb + f4 * 4);
            __bf16* dst = &As[m][f4 * 4];
            dst[0] = (__bf16)(xv.x - bd.x);
            dst[1] = (__bf16)(xv.y - bd.y);
            dst[2] = (__bf16)(xv.z - bd.z);
            dst[3] = (__bf16)(xv.w - bd.w);
        }
        // ---- cooperative load B chunk: W_enc[t][nBase+n][kb+k] -> Bs[k][n] (transposed)
#pragma unroll
        for (int r = 0; r < 2; ++r) {
            int lin = tid + r * 256;          // 0..511
            int n   = lin >> 3;               // 0..63
            int f4  = lin & 7;
            const float4 wv = *(const float4*)(W_enc + ((size_t)t * DSAE + nBase + n) * DIN + kb + f4 * 4);
            Bs[f4 * 4 + 0][n] = (__bf16)wv.x;
            Bs[f4 * 4 + 1][n] = (__bf16)wv.y;
            Bs[f4 * 4 + 2][n] = (__bf16)wv.z;
            Bs[f4 * 4 + 3][n] = (__bf16)wv.w;
        }
        __syncthreads();

        // B fragment: lane&15 = N col; elems = K {0..7,16..23} or {8..15,24..31}
        v16bf bfrag;
#pragma unroll
        for (int s = 0; s < 8; ++s) {
            bfrag[s]     = Bs[(lane >> 4) * 8 + s][nSub + (lane & 15)];
            bfrag[s + 8] = Bs[16 + (lane >> 4) * 8 + s][nSub + (lane & 15)];
        }

        const int khalf = (lane >> 4) * 8;    // K half-split for A operand
#pragma unroll
        for (int r2 = 0; r2 < 4; ++r2) {
            // A fragment: lane&15 = M row; elems = K {0..7,16..23} or {8..15,24..31}
            const __bf16* ap = &As[mWave + 16 * r2 + (lane & 15)][khalf];
            v16bf afrag = concat8(*(const v8bf*)ap, *(const v8bf*)(ap + 16));
            acc[r2] = __builtin_amdgcn_wmma_f32_16x16x32_bf16(
                false, afrag, false, bfrag, (short)0, acc[r2], false, false);
        }
        __syncthreads();
    }

    // ---- epilogue: C layout  VGPR r -> M = r (lanes 0-15) / 8+r (lanes 16-31), N = lane&15
    const int   nCol  = nBase + nSub + (lane & 15);
    const int   mHalf = (lane >> 4) * 8;
    const float be    = b_enc[(size_t)t * DSAE + nCol];
#pragma unroll
    for (int r2 = 0; r2 < 4; ++r2) {
#pragma unroll
        for (int r = 0; r < 8; ++r) {
            int gM = mBase + mWave + 16 * r2 + mHalf + r;
            pre[((size_t)gM * T_ + t) * DSAE + nCol] = acc[r2][r] + be;
        }
    }
}

// ---------------------------------------------------------------------------
// Kernel 2: per-row top-32 selection (in place on the u region).
// One block per (b,t) row. Row staged in LDS; kth-largest via binary search
// on the value range; ties resolved with a shared take-counter.
// ---------------------------------------------------------------------------
__global__ __launch_bounds__(256)
void sae_topk(float* __restrict__ u, float* __restrict__ lossSlot) {
    __shared__ float sv[DSAE];       // 32 KB row cache
    __shared__ float sredf[256];
    __shared__ int   sredi[256];
    __shared__ int   sTake;

    const int tid = threadIdx.x;
    const int row = blockIdx.x;
    float* rowp = u + (size_t)row * DSAE;

    if (row == 0 && tid == 0) lossSlot[0] = 0.0f;  // decoder atomics come later in-stream

    float mx = -3.4e38f, mn = 3.4e38f;
    for (int i = tid; i < DSAE; i += 256) {
        float v = rowp[i];
        sv[i] = v;
        mx = fmaxf(mx, v);
        mn = fminf(mn, v);
    }
    sredf[tid] = mx; __syncthreads();
    for (int s = 128; s > 0; s >>= 1) { if (tid < s) sredf[tid] = fmaxf(sredf[tid], sredf[tid + s]); __syncthreads(); }
    mx = sredf[0]; __syncthreads();
    sredf[tid] = mn; __syncthreads();
    for (int s = 128; s > 0; s >>= 1) { if (tid < s) sredf[tid] = fminf(sredf[tid], sredf[tid + s]); __syncthreads(); }
    mn = sredf[0]; __syncthreads();

    // invariant: count(> lo) >= K, count(> hi) < K
    float lo = mn - 1.0f, hi = mx;
    for (int it = 0; it < 40; ++it) {
        float mid = 0.5f * (lo + hi);
        int c = 0;
        for (int i = tid; i < DSAE; i += 256) c += (sv[i] > mid) ? 1 : 0;
        sredi[tid] = c; __syncthreads();
        for (int s = 128; s > 0; s >>= 1) { if (tid < s) sredi[tid] += sredi[tid + s]; __syncthreads(); }
        int tot = sredi[0]; __syncthreads();
        if (tot >= K_) lo = mid; else hi = mid;
    }

    int c = 0;
    for (int i = tid; i < DSAE; i += 256) c += (sv[i] > hi) ? 1 : 0;
    sredi[tid] = c; __syncthreads();
    for (int s = 128; s > 0; s >>= 1) { if (tid < s) sredi[tid] += sredi[tid + s]; __syncthreads(); }
    const int cHi = sredi[0];
    if (tid == 0) sTake = 0;
    __syncthreads();
    const int extra = K_ - cHi;   // >= 1 by invariant

    for (int i = tid; i < DSAE; i += 256) {
        float v = sv[i];
        bool sel;
        if (v > hi)       sel = true;
        else if (v > lo)  { int p = atomicAdd(&sTake, 1); sel = (p < extra); }
        else              sel = false;
        rowp[i] = sel ? fmaxf(v, 0.0f) : 0.0f;   // relu(topk) scattered, zeros elsewhere
    }
}

// ---------------------------------------------------------------------------
// Kernel 3: sparse decode + loss. One block per (b,t) row; gathers the <=32
// active latents straight out of u, accumulates x_hat over 768 dims
// (3 per thread), and atomically accumulates mean row SSE into loss.
// ---------------------------------------------------------------------------
__global__ __launch_bounds__(256)
void sae_decode(const float* __restrict__ x,
                const float* __restrict__ W_dec,
                const float* __restrict__ b_dec,
                const float* __restrict__ u,
                float* __restrict__ xhat,
                float* __restrict__ loss) {
    __shared__ int   sIdx[64];
    __shared__ float sVal[64];
    __shared__ int   sCnt;
    __shared__ float sred[256];

    const int tid = threadIdx.x;
    const int row = blockIdx.x;
    const int t   = row & (T_ - 1);

    if (tid == 0) sCnt = 0;
    __syncthreads();

    const float* urow = u + (size_t)row * DSAE;
    for (int i = tid; i < DSAE; i += 256) {
        float v = urow[i];
        if (v > 0.0f) {
            int p = atomicAdd(&sCnt, 1);
            if (p < 64) { sIdx[p] = i; sVal[p] = v; }
        }
    }
    __syncthreads();
    const int cnt = (sCnt < 64) ? sCnt : 64;   // <= 32 in practice

    const int d0 = tid, d1 = tid + 256, d2 = tid + 512;  // DIN = 768 = 3*256
    float a0 = b_dec[(size_t)t * DIN + d0];
    float a1 = b_dec[(size_t)t * DIN + d1];
    float a2 = b_dec[(size_t)t * DIN + d2];
    const float* Wt = W_dec + (size_t)t * DIN * DSAE;
    for (int j = 0; j < cnt; ++j) {
        const float vv = sVal[j];
        const int   h  = sIdx[j];
        a0 += vv * Wt[(size_t)d0 * DSAE + h];
        a1 += vv * Wt[(size_t)d1 * DSAE + h];
        a2 += vv * Wt[(size_t)d2 * DSAE + h];
    }
    const size_t o = (size_t)row * DIN;
    xhat[o + d0] = a0;
    xhat[o + d1] = a1;
    xhat[o + d2] = a2;

    const float e0 = a0 - x[o + d0];
    const float e1 = a1 - x[o + d1];
    const float e2 = a2 - x[o + d2];
    sred[tid] = e0 * e0 + e1 * e1 + e2 * e2;
    __syncthreads();
    for (int s = 128; s > 0; s >>= 1) { if (tid < s) sred[tid] += sred[tid + s]; __syncthreads(); }
    if (tid == 0) atomicAdd(loss, sred[0] * (1.0f / (float)(B_ * T_)));
}

// ---------------------------------------------------------------------------
extern "C" void kernel_launch(void* const* d_in, const int* in_sizes, int n_in,
                              void* d_out, int out_size, void* d_ws, size_t ws_size,
                              hipStream_t stream) {
    (void)in_sizes; (void)n_in; (void)d_ws; (void)ws_size; (void)out_size;
    const float* x     = (const float*)d_in[0];
    const float* W_enc = (const float*)d_in[1];
    const float* b_enc = (const float*)d_in[2];
    const float* W_dec = (const float*)d_in[3];
    const float* b_dec = (const float*)d_in[4];
    // d_in[5] = k (always 32 here)

    float* out   = (float*)d_out;
    float* lossP = out;                                   // 1 element
    float* xhatP = out + 1;                               // B*T*DIN
    float* uP    = out + 1 + (size_t)B_ * T_ * DIN;       // B*T*DSAE

    // Encoder: grid = (N tiles of 64, M tiles of 128, T)
    dim3 g1(DSAE / 64, B_ / 128, T_);
    sae_encode_wmma<<<g1, 256, 0, stream>>>(x, W_enc, b_enc, b_dec, uP);

    // Top-K in place on u (also zeroes the loss slot)
    sae_topk<<<dim3(B_ * T_), 256, 0, stream>>>(uP, lossP);

    // Sparse decode + x_hat + loss
    sae_decode<<<dim3(B_ * T_), 256, 0, stream>>>(x, W_dec, b_dec, uP, xhatP, lossP);
}
